// OptimizeLSTM_36833639530916
// MI455X (gfx1250) — matrix operand order repaired and checked by reading, not compiled, not run on hardware
//
#include <hip/hip_runtime.h>
#include <stdint.h>

// Problem dims (fixed by the reference)
#define BB 64
#define TT 512
#define FF 512
#define HH 1024
#define GG 4096   // 4*H

// Scan decomposition
#define NWG   32          // persistent workgroups; each owns 32 h-columns
#define LCOLS 128         // gate columns per WG (4 strips x 32)
#define LDSK  1032        // padded K stride (1024 + 8) -> 516 dwords: bank-conflict-free
#define LDS_BYTES (LCOLS * LDSK * 2)   // 264,192 B  (< 320 KB WGP LDS)

typedef __bf16 bf16_t;
typedef bf16_t v16bf __attribute__((ext_vector_type(16)));
typedef float  v8f   __attribute__((ext_vector_type(8)));

// 32-byte WMMA operand fragment; filled via two 16B loads.
union Frag {
  v16bf v;
  uint4 q[2];
};

static __device__ __forceinline__ unsigned short f2bf(float f) {
  unsigned u = __float_as_uint(f);
  u += 0x7FFFu + ((u >> 16) & 1u);           // round-to-nearest-even
  return (unsigned short)(u >> 16);
}
static __device__ __forceinline__ float bf2f(unsigned short b) {
  return __uint_as_float(((unsigned)b) << 16);
}
static __device__ __forceinline__ float fast_sigmoid(float x) {
  float e = __builtin_amdgcn_exp2f(-x * 1.44269504088896340736f);
  return __builtin_amdgcn_rcpf(1.0f + e);
}
static __device__ __forceinline__ float fast_tanh(float x) {
  // tanh(x) = 1 - 2/(exp2(2x*log2e)+1); saturates correctly at +-1
  float e = __builtin_amdgcn_exp2f(x * 2.88539008177792681472f);
  return 1.0f - 2.0f * __builtin_amdgcn_rcpf(e + 1.0f);
}

// ---------------------------------------------------------------- prep kernels

__global__ void cast_f32_to_bf16(const float* __restrict__ src,
                                 unsigned short* __restrict__ dst, int n) {
  int stride = gridDim.x * blockDim.x;
  for (int i = blockIdx.x * blockDim.x + threadIdx.x; i < n; i += stride)
    dst[i] = f2bf(src[i]);
}

// src [R][C] f32  ->  dst [C][R] bf16  (write-coalesced)
__global__ void transpose_cast(const float* __restrict__ src,
                               unsigned short* __restrict__ dst, int R, int C) {
  int n = R * C;
  int stride = gridDim.x * blockDim.x;
  for (int i = blockIdx.x * blockDim.x + threadIdx.x; i < n; i += stride) {
    int c = i / R;
    int r = i - c * R;
    dst[i] = f2bf(src[(size_t)r * C + c]);
  }
}

__global__ void zero_u32(unsigned* __restrict__ p, int n) {
  int stride = gridDim.x * blockDim.x;
  for (int i = blockIdx.x * blockDim.x + threadIdx.x; i < n; i += stride)
    p[i] = 0u;
}

// ------------------------------------------------- input projection xU = x@U+b
// A = x_bf16 [B*T][F], B = U^T bf16 [G][F], out xu bf16 [T][B][G]
// wave tile: 16 rows x 64 cols; 8 waves/block.
__global__ __launch_bounds__(256)
void proj_gemm(const unsigned short* __restrict__ xb,
               const unsigned short* __restrict__ Ut,
               const float* __restrict__ bias,
               unsigned short* __restrict__ xu) {
  const int lane = threadIdx.x & 31;
  const int wave = threadIdx.x >> 5;
  const int gid  = blockIdx.x * 8 + wave;
  const int mtile = gid >> 6;           // 0..2047
  const int cg    = gid & 63;           // 64-column group
  const int half  = lane >> 4;
  const int l16   = lane & 15;
  const int row   = (mtile << 4) + l16; // A row = b*T + t

  const unsigned short* arow = xb + (size_t)row * FF;
  v8f acc[4] = {};

  for (int kb = 0; kb < FF; kb += 32) {
    Frag a;
    a.q[0] = *(const uint4*)(arow + kb + 8 * half);        // K 8h .. 8h+7
    a.q[1] = *(const uint4*)(arow + kb + 16 + 8 * half);   // K 16+8h .. +7
#pragma unroll
    for (int nt = 0; nt < 4; ++nt) {
      const int col = (cg << 6) + (nt << 4) + l16;
      const unsigned short* bp = Ut + (size_t)col * FF + kb;
      Frag bf;
      bf.q[0] = *(const uint4*)(bp + 8 * half);
      bf.q[1] = *(const uint4*)(bp + 16 + 8 * half);
      acc[nt] = __builtin_amdgcn_wmma_f32_16x16x32_bf16(
          false, a.v, false, bf.v, (short)0, acc[nt], false, false);
    }
  }

  const int r0 = mtile << 4;
#pragma unroll
  for (int nt = 0; nt < 4; ++nt) {
    const int col = (cg << 6) + (nt << 4) + l16;
    const float bb = bias[col];
#pragma unroll
    for (int v = 0; v < 8; ++v) {
      const int r = r0 + v + 8 * half;   // C/D layout: M = v + 8*half
      const int b = r >> 9;              // r / T
      const int t = r & 511;             // r % T
      xu[((size_t)t * BB + b) * GG + col] = f2bf(acc[nt][v] + bb);
    }
  }
}

// -------------------------------------------------------------- recurrent scan
// Persistent: NWG=32 WGs x 256 threads on 32 WGPs. WG wg owns h-columns
// [wg*32, wg*32+32) and gate columns {s*H + those}. Its 256KB W^T slice lives
// in LDS for the entire scan (padded, bank-conflict-free), so the per-step
// inner loop is WMMA fed by ds_load_b128 (B) + pipelined global A loads.
// Cell state c stays in registers for all 512 steps; h ping-pongs in ws bf16.
__global__ __launch_bounds__(256)
void lstm_scan(const unsigned short* __restrict__ xu,  // [T][B][G] bf16
               const unsigned short* __restrict__ Wt,  // [G][H]   bf16 (W^T)
               unsigned short* __restrict__ h0,        // [B][H] bf16 ping
               unsigned short* __restrict__ h1,        // [B][H] bf16 pong
               float* __restrict__ out,                // [B][T][H]
               float* __restrict__ hT,                 // [B][H]
               float* __restrict__ cT,                 // [B][H]
               unsigned* __restrict__ bar) {
  extern __shared__ unsigned short wlds[];   // [LCOLS][LDSK]

  const int wg   = blockIdx.x;        // 0..31
  const int lane = threadIdx.x & 31;
  const int wave = threadIdx.x >> 5;  // 0..7
  const int half = lane >> 4;
  const int l16  = lane & 15;
  const int nn    = wave & 1;         // column sub-tile within each gate strip
  const int mtile = wave >> 1;        // 0..3 -> rows [mtile*16, +16)
  const int jcol  = (wg << 5) + (nn << 4) + l16;   // owned h column, 0..1023
  const int rowA  = (mtile << 4) + l16;

  // ---- one-time W slice preload into LDS (amortized over 512 steps)
  for (int c = threadIdx.x; c < LCOLS * 64; c += 256) {
    const int lc   = c >> 6;                 // local gate column 0..127
    const int k16  = (c & 63) << 4;          // K offset, 16-elem chunks
    const int gcol = (lc >> 5) * HH + (wg << 5) + (lc & 31);
    const unsigned short* gp = Wt + (size_t)gcol * HH + k16;
    *(uint4*)(wlds + lc * LDSK + k16)     = *(const uint4*)(gp);
    *(uint4*)(wlds + lc * LDSK + k16 + 8) = *(const uint4*)(gp + 8);
  }
  __syncthreads();

  float creg[8];
#pragma unroll
  for (int v = 0; v < 8; ++v) creg[v] = 0.0f;

  for (int t = 0; t < TT; ++t) {
    const unsigned short* hb = (t & 1) ? h1 : h0;  // read h_{t}
    unsigned short*       hn = (t & 1) ? h0 : h1;  // write h_{t+1}
    const unsigned short* ar = hb + (size_t)rowA * HH;

    // gate inputs are h-independent: issue these global loads before the GEMM
    float xg[4][8];
#pragma unroll
    for (int v = 0; v < 8; ++v) {
      const int b = (mtile << 4) + v + 8 * half;
      const unsigned short* xrow = xu + ((size_t)t * BB + b) * GG;
#pragma unroll
      for (int s = 0; s < 4; ++s) xg[s][v] = bf2f(xrow[s * HH + jcol]);
    }

    v8f acc[4] = {};

    // software-pipelined k loop: A from global (pipelined), B from LDS
    Frag a_cur, a_nxt;
    a_cur.q[0] = *(const uint4*)(ar + 8 * half);
    a_cur.q[1] = *(const uint4*)(ar + 16 + 8 * half);
#pragma unroll 4
    for (int kb = 0; kb < HH; kb += 32) {
      if (kb + 32 < HH) {
        a_nxt.q[0] = *(const uint4*)(ar + kb + 32 + 8 * half);
        a_nxt.q[1] = *(const uint4*)(ar + kb + 48 + 8 * half);
      }
#pragma unroll
      for (int s = 0; s < 4; ++s) {
        const unsigned short* bp =
            wlds + (s * 32 + (nn << 4) + l16) * LDSK + kb;
        Frag bf;
        bf.q[0] = *(const uint4*)(bp + 8 * half);       // ds_load_b128, conflict-free
        bf.q[1] = *(const uint4*)(bp + 16 + 8 * half);
        acc[s] = __builtin_amdgcn_wmma_f32_16x16x32_bf16(
            false, a_cur.v, false, bf.v, (short)0, acc[s], false, false);
      }
      a_cur = a_nxt;
    }

    // gate epilogue: f,i,o,c~ -> c,h (c stays in registers)
#pragma unroll
    for (int v = 0; v < 8; ++v) {
      const int b = (mtile << 4) + v + 8 * half;
      float g0 = acc[0][v] + xg[0][v];
      float g1 = acc[1][v] + xg[1][v];
      float g2 = acc[2][v] + xg[2][v];
      float g3 = acc[3][v] + xg[3][v];
      float fg  = fast_sigmoid(g0);
      float ig  = fast_sigmoid(g1);
      float og  = fast_sigmoid(g2);
      float cg_ = fast_sigmoid(g3);   // reference uses sigmoid for candidate
      float cc = fg * creg[v] + ig * cg_;
      creg[v] = cc;
      float hh = og * fast_tanh(cc);
      out[((size_t)b * TT + t) * HH + jcol] = hh;
      hn[(size_t)b * HH + jcol] = f2bf(hh);
      if (t == TT - 1) {
        hT[(size_t)b * HH + jcol] = hh;
        cT[(size_t)b * HH + jcol] = cc;
      }
    }

    if (t + 1 < TT) {
      // warm L2 for next step's gate inputs (global_prefetch_b8);
      // 8192 threads x 64B covers the whole 512KB xu[t+1] block
      __builtin_prefetch(xu + ((size_t)(t + 1) * BB) * GG +
                             (size_t)(wg * 256 + threadIdx.x) * 32, 0, 1);
    }

    // ---- step boundary: grid barrier across the NWG persistent workgroups
    __threadfence();
    __syncthreads();
    if (threadIdx.x == 0) {
      __hip_atomic_fetch_add(bar, 1u, __ATOMIC_RELEASE, __HIP_MEMORY_SCOPE_AGENT);
      const unsigned target = (unsigned)NWG * (unsigned)(t + 1);
      while (__hip_atomic_load(bar, __ATOMIC_ACQUIRE, __HIP_MEMORY_SCOPE_AGENT) < target)
        __builtin_amdgcn_s_sleep(1);
    }
    if (wave == 0)
      __builtin_amdgcn_s_cluster_barrier();  // s_barrier_signal/-wait -3; NOP if unclustered
    __syncthreads();
  }
}

// ------------------------------------------------------------------- launcher

extern "C" void kernel_launch(void* const* d_in, const int* in_sizes, int n_in,
                              void* d_out, int out_size, void* d_ws, size_t ws_size,
                              hipStream_t stream) {
  (void)in_sizes; (void)n_in; (void)out_size; (void)ws_size;

  const float* x    = (const float*)d_in[0];   // [B,T,F]
  const float* U    = (const float*)d_in[1];   // [F,4H]
  const float* W    = (const float*)d_in[2];   // [H,4H]
  const float* bias = (const float*)d_in[3];   // [4H]

  // workspace layout (~315 MB total, all offsets 256B-aligned)
  uint8_t* ws = (uint8_t*)d_ws;
  const size_t XU_OFF  = 0;                                    // 268,435,456 B
  const size_t XB_OFF  = XU_OFF + (size_t)TT * BB * GG * 2;    //  33,554,432 B
  const size_t UT_OFF  = XB_OFF + (size_t)BB * TT * FF * 2;    //   4,194,304 B
  const size_t WT_OFF  = UT_OFF + (size_t)GG * FF * 2;         //   8,388,608 B
  const size_t H0_OFF  = WT_OFF + (size_t)GG * HH * 2;         //     131,072 B
  const size_t H1_OFF  = H0_OFF + (size_t)BB * HH * 2;         //     131,072 B
  const size_t BAR_OFF = H1_OFF + (size_t)BB * HH * 2;         //         256 B

  unsigned short* xu = (unsigned short*)(ws + XU_OFF);
  unsigned short* xb = (unsigned short*)(ws + XB_OFF);
  unsigned short* Ut = (unsigned short*)(ws + UT_OFF);
  unsigned short* Wt = (unsigned short*)(ws + WT_OFF);
  unsigned short* h0 = (unsigned short*)(ws + H0_OFF);
  unsigned short* h1 = (unsigned short*)(ws + H1_OFF);
  unsigned*       bar = (unsigned*)(ws + BAR_OFF);

  float* out = (float*)d_out;                        // hidden_seq [B,T,H]
  float* hT  = out + (size_t)BB * TT * HH;           // h_T [B,H]
  float* cT  = hT + (size_t)BB * HH;                 // c_T [B,H]

  // allow 264KB dynamic LDS for the scan kernel (WGP has 320KB)
  (void)hipFuncSetAttribute((const void*)lstm_scan,
                            hipFuncAttributeMaxDynamicSharedMemorySize,
                            LDS_BYTES);

  cast_f32_to_bf16<<<2048, 256, 0, stream>>>(x, xb, BB * TT * FF);
  transpose_cast  <<<2048, 256, 0, stream>>>(U, Ut, FF, GG);   // -> [4096][512]
  transpose_cast  <<<2048, 256, 0, stream>>>(W, Wt, HH, GG);   // -> [4096][1024]
  // zero h ping-pong + barrier counter (contiguous region)
  zero_u32<<<64, 256, 0, stream>>>((unsigned*)(ws + H0_OFF),
                                   (int)(((size_t)BB * HH * 2 * 2) / 4 + 64));

  proj_gemm<<<16384, 256, 0, stream>>>(xb, Ut, bias, xu);
  lstm_scan<<<NWG, 256, LDS_BYTES, stream>>>(xu, Wt, h0, h1, out, hT, cT, bar);
}
